// EDCM_30666066493716
// MI455X (gfx1250) — matrix operand order repaired and checked by reading, not compiled, not run on hardware
//
#include <hip/hip_runtime.h>
#include <hip/hip_bf16.h>
#include <math.h>

// ---------------- problem constants ----------------
#define B_    16
#define CIN   256
#define COUT  256
#define H_    64
#define W_    64
#define HW    4096           // 64*64
#define KPOSN 9              // 3x3 taps
#define KTOT  (KPOSN * CIN)  // 2304, GEMM K
#define MIDN  64
#define BN_EPS 1e-5f

// ---------------- WMMA tile config -----------------
#define MT 128    // block tile: output channels
#define NT 256    // block tile: pixels
#define KC 32     // K chunk per WMMA step (bf16 16x16x32)
#define NSTEPS (KTOT / KC)       // 72
#define LDS_STRIDE 40            // ushorts per row (32 data + 8 pad = 80B, 16B aligned)

typedef __attribute__((ext_vector_type(16))) __bf16        v16bf;
typedef __attribute__((ext_vector_type(8)))  float         v8f;
typedef __attribute__((ext_vector_type(8)))  unsigned int  v8u;
typedef __attribute__((address_space(3)))    unsigned short lds_us;

// float -> bf16 (round to nearest even), as raw u16
static __device__ __forceinline__ unsigned short f2bf(float f) {
    unsigned int u = __float_as_uint(f);
    u += 0x7FFFu + ((u >> 16) & 1u);
    return (unsigned short)(u >> 16);
}

// 32-byte async copy global->LDS (2x b128, ASYNCcnt-tracked, no VGPR round trip).
// INST_OFFSET is added to BOTH the LDS and global addresses per the CDNA5 async
// addressing rule, so one vdst/vaddr pair stages 32 contiguous bytes.
static __device__ __forceinline__ void async_cp32(unsigned short* lds_generic,
                                                  const unsigned short* gbase,
                                                  unsigned int voff_bytes) {
    lds_us* l = (lds_us*)lds_generic;   // addrspacecast: 32-bit LDS byte address
    asm volatile(
        "global_load_async_to_lds_b128 %0, %1, %2 offset:0\n\t"
        "global_load_async_to_lds_b128 %0, %1, %2 offset:16"
        :
        : "v"(l), "v"(voff_bytes), "s"(gbase)
        : "memory");
}

static __device__ __forceinline__ void wait_async0() {
    asm volatile("s_wait_asynccnt 0x0" ::: "memory");
}

// ================= 1) global average pool =================
__global__ __launch_bounds__(256) void gap_kernel(const float* __restrict__ x,
                                                  float* __restrict__ pooled) {
    __shared__ float red[256];
    const int bc  = blockIdx.x;               // 0 .. B_*CIN-1
    const int tid = threadIdx.x;
    const float* row = x + (size_t)bc * HW;
    float s = 0.f;
    #pragma unroll
    for (int i = 0; i < HW / 256; ++i) s += row[tid + i * 256];
    red[tid] = s;
    __syncthreads();
    for (int off = 128; off > 0; off >>= 1) {
        if (tid < off) red[tid] += red[tid + off];
        __syncthreads();
    }
    if (tid == 0) pooled[bc] = red[0] * (1.0f / HW);
}

// ================= 2) attention branches -> alpha =================
__global__ __launch_bounds__(64) void alpha_kernel(
    const float* __restrict__ pooled,
    const float* __restrict__ w1s, const float* __restrict__ w2s, const float* __restrict__ b2s,
    const float* __restrict__ w1c, const float* __restrict__ w2c, const float* __restrict__ b2c,
    const float* __restrict__ w1f, const float* __restrict__ w2f, const float* __restrict__ b2f,
    const float* __restrict__ w1w, const float* __restrict__ w2w, const float* __restrict__ b2w,
    float* __restrict__ alpha) {
    __shared__ float h[MIDN];
    __shared__ float logit[4];
    __shared__ float prod[4];
    const int b = blockIdx.x;
    const int t = threadIdx.x;
    const float* W1[4] = {w1s, w1c, w1f, w1w};
    const float* W2[4] = {w2s, w2c, w2f, w2w};
    const float* B2[4] = {b2s, b2c, b2f, b2w};
    if (t < 4) prod[t] = 1.0f;
    __syncthreads();
    for (int br = 0; br < 4; ++br) {
        float s = 0.f;
        const float* p = pooled + b * CIN;
        const float* w = W1[br] + t * CIN;
        #pragma unroll 8
        for (int c = 0; c < CIN; ++c) s += p[c] * w[c];
        h[t] = fmaxf(s, 0.f);
        __syncthreads();
        if (t < 4) {
            float l = B2[br][t];
            const float* w2 = W2[br] + t * MIDN;
            #pragma unroll 8
            for (int m = 0; m < MIDN; ++m) l += h[m] * w2[m];
            logit[t] = l;
        }
        __syncthreads();
        if (t == 0) {
            float mx = fmaxf(fmaxf(logit[0], logit[1]), fmaxf(logit[2], logit[3]));
            float e0 = __expf(logit[0] - mx), e1 = __expf(logit[1] - mx);
            float e2 = __expf(logit[2] - mx), e3 = __expf(logit[3] - mx);
            float inv = 1.0f / (e0 + e1 + e2 + e3);
            prod[0] *= e0 * inv; prod[1] *= e1 * inv;
            prod[2] *= e2 * inv; prod[3] *= e3 * inv;
        }
        __syncthreads();
    }
    if (t < 4) alpha[b * 4 + t] = prod[t];
}

// ================= 3) dynamic weights (bf16, GEMM-A layout) =================
// dynw[b][co][kpos*CIN + ci] = bf16( sum_n alpha[b][n] * weight[n][co][ci][kpos] )
__global__ __launch_bounds__(256) void dynw_kernel(const float* __restrict__ weight,
                                                   const float* __restrict__ alpha,
                                                   unsigned short* __restrict__ dynw) {
    const long long idx = (long long)blockIdx.x * 256 + threadIdx.x;
    const int ci   = (int)(idx & (CIN - 1));
    const int kpos = (int)((idx >> 8) % KPOSN);
    const int co   = (int)((idx / (KPOSN * CIN)) & (COUT - 1));
    const int b    = (int)(idx / ((long long)COUT * KPOSN * CIN));
    const float* a = alpha + b * 4;
    float v = 0.f;
    #pragma unroll
    for (int n = 0; n < 4; ++n) {
        size_t wi = (((size_t)n * COUT + co) * CIN + ci) * KPOSN + kpos;
        v += a[n] * weight[wi];
    }
    dynw[idx] = f2bf(v);
}

// ================= 4) x -> NHWC bf16 (GEMM-B layout) =================
__global__ __launch_bounds__(256) void xcvt_kernel(const float* __restrict__ x,
                                                   unsigned short* __restrict__ xb) {
    __shared__ unsigned short tile[32][33];
    const int p0  = blockIdx.x * 32;
    const int ci0 = blockIdx.y * 32;
    const int b   = blockIdx.z;
    const int t   = threadIdx.x;
    #pragma unroll
    for (int k = 0; k < 4; ++k) {
        int e = t + k * 256;
        int r = e >> 5, c = e & 31;                 // r = ci_local, c = p_local
        tile[r][c] = f2bf(x[((size_t)(b * CIN + ci0 + r)) * HW + p0 + c]);
    }
    __syncthreads();
    #pragma unroll
    for (int k = 0; k < 4; ++k) {
        int e = t + k * 256;
        int r = e >> 5, c = e & 31;                 // r = p_local, c = ci_local
        xb[((size_t)b * HW + p0 + r) * CIN + ci0 + c] = tile[c][r];
    }
}

// ================= 5) implicit-GEMM conv via bf16 WMMA =================
// Double-buffered async pipeline: tile k+1 streams into LDS (ASYNCcnt) while
// the 16 WMMAs of tile k run. Block tile 128co x 256px, wave tile 64x64.
__global__ __launch_bounds__(256) void conv_wmma_kernel(
    const unsigned short* __restrict__ dynw,  // [B][COUT][KTOT] bf16
    const unsigned short* __restrict__ xb,    // [B][HW][CIN]    bf16
    float* __restrict__ out) {                // [B][COUT][HW]   f32
    __shared__ __align__(16) unsigned short Asm[2][MT * LDS_STRIDE];  // 2 x 10KB
    __shared__ __align__(16) unsigned short Bsm[2][NT * LDS_STRIDE];  // 2 x 20KB

    const int px0 = blockIdx.x * NT;
    const int co0 = blockIdx.y * MT;
    const int b   = blockIdx.z;

    const int tid  = threadIdx.x;
    const int wave = tid >> 5;          // 0..7
    const int lane = tid & 31;
    const int wm   = wave >> 2;         // 0..1 : co offset wm*64
    const int wn   = wave & 3;          // 0..3 : px offset wn*64

    v8f acc[4][4];
    #pragma unroll
    for (int i = 0; i < 4; ++i)
        #pragma unroll
        for (int j = 0; j < 4; ++j)
            #pragma unroll
            for (int r = 0; r < 8; ++r) acc[i][j][r] = 0.0f;

    const unsigned short* dwB = dynw + ((size_t)b * COUT + co0) * KTOT;
    const unsigned short* xbB = xb + (size_t)b * HW * CIN;

    // staging geometry: 32B chunks; A = 256 chunks (1/thread), B = 512 (2/thread)
    const int arow  = tid >> 1;   // 0..127
    const int ahalf = tid & 1;

    // per-lane fragment geometry (CDNA5 ISA 16-bit WMMA VGPR layouts)
    const int frow  = lane & 15;
    const int akb   = (lane >> 4) * 8;    // A: K bytes [2*akb,+16) and [2*akb+32,+16)
    const int bkoff = (lane >> 4) * 8;    // B: K [bkoff,+8) and [bkoff+16,+8)

    // ---- tile stager (async copies; OOB pixels -> ds_store zeros) ----
    auto stage = [&](int kt, int p) {
        const int kpos = kt >> 3;             // 0..8
        const int cb   = (kt & 7) * KC;       // 0..224
        // A tile: 128 co x 32 k
        {
            unsigned int voff = (unsigned int)arow * (KTOT * 2) +
                                (unsigned int)(kpos * CIN + cb + ahalf * 16) * 2;
            async_cp32(&Asm[p][arow * LDS_STRIDE + ahalf * 16], dwB, voff);
        }
        // B tile: 256 px x 32 k, SAME padding
        const int dy = kpos / 3 - 1, dx = kpos % 3 - 1;
        #pragma unroll
        for (int c = 0; c < 2; ++c) {
            const int chunk = tid + c * 256;      // 0..511
            const int row   = chunk >> 1;         // 0..255
            const int half  = chunk & 1;
            const int pix   = px0 + row;
            const int yy = (pix >> 6) + dy, xx = (pix & 63) + dx;
            unsigned short* dst = &Bsm[p][row * LDS_STRIDE + half * 16];
            if ((unsigned)yy < (unsigned)H_ && (unsigned)xx < (unsigned)W_) {
                unsigned int voff = (unsigned int)(yy * W_ + xx) * (CIN * 2) +
                                    (unsigned int)(cb + half * 16) * 2;
                async_cp32(dst, xbB, voff);
            } else {
                uint4 z = {0, 0, 0, 0};
                ((uint4*)dst)[0] = z;
                ((uint4*)dst)[1] = z;
            }
        }
    };

    stage(0, 0);                      // prologue

    #pragma unroll 1
    for (int kt = 0; kt < NSTEPS; ++kt) {
        const int p = kt & 1;
        wait_async0();                // own async copies for buffer p done
        __syncthreads();              // all waves' staging of buffer p done
        if (kt + 1 < NSTEPS) stage(kt + 1, p ^ 1);   // overlap with compute

        // ---- fragments from LDS ----
        v16bf bfrag[4];
        #pragma unroll
        for (int j = 0; j < 4; ++j) {
            const unsigned short* q =
                &Bsm[p][(wn * 64 + j * 16 + frow) * LDS_STRIDE + bkoff];
            uint4 lo = ((const uint4*)q)[0];
            uint4 hi = ((const uint4*)(q + 16))[0];
            v8u t = {lo.x, lo.y, lo.z, lo.w, hi.x, hi.y, hi.z, hi.w};
            bfrag[j] = __builtin_bit_cast(v16bf, t);
        }
        v16bf afrag[4];
        #pragma unroll
        for (int i = 0; i < 4; ++i) {
            const unsigned short* q =
                &Asm[p][(wm * 64 + i * 16 + frow) * LDS_STRIDE + akb];
            uint4 lo = ((const uint4*)q)[0];        // K: akb..akb+7
            uint4 hi = ((const uint4*)(q + 16))[0]; // K: akb+16..akb+23
            v8u t = {lo.x, lo.y, lo.z, lo.w, hi.x, hi.y, hi.z, hi.w};
            afrag[i] = __builtin_bit_cast(v16bf, t);
        }
        #pragma unroll
        for (int i = 0; i < 4; ++i)
            #pragma unroll
            for (int j = 0; j < 4; ++j)
                acc[i][j] = __builtin_amdgcn_wmma_f32_16x16x32_bf16(
                    false, afrag[i], false, bfrag[j],
                    (short)0, acc[i][j], false, false);
    }

    // ---- write D (C/D layout: col = lane&15, row = r + 8*(lane>>4)) ----
    #pragma unroll
    for (int i = 0; i < 4; ++i) {
        #pragma unroll
        for (int j = 0; j < 4; ++j) {
            const int n     = px0 + wn * 64 + j * 16 + (lane & 15);
            const int mbase = co0 + wm * 64 + i * 16 + (lane >> 4) * 8;
            #pragma unroll
            for (int r = 0; r < 8; ++r)
                out[((size_t)b * COUT + (mbase + r)) * HW + n] = acc[i][j][r];
        }
    }
}

// ================= 6) BN batch statistics =================
__global__ __launch_bounds__(256) void bnstats_kernel(const float* __restrict__ out,
                                                      float* __restrict__ mean,
                                                      float* __restrict__ rstd) {
    __shared__ float rs[256], rq[256];
    const int co  = blockIdx.x;
    const int tid = threadIdx.x;
    float s = 0.f, q = 0.f;
    for (int i = tid; i < B_ * HW; i += 256) {
        int bb = i >> 12, p = i & (HW - 1);
        float v = out[((size_t)bb * COUT + co) * HW + p];
        s += v; q += v * v;
    }
    rs[tid] = s; rq[tid] = q;
    __syncthreads();
    for (int off = 128; off > 0; off >>= 1) {
        if (tid < off) { rs[tid] += rs[tid + off]; rq[tid] += rq[tid + off]; }
        __syncthreads();
    }
    if (tid == 0) {
        const float inv = 1.0f / (B_ * HW);
        float m = rs[0] * inv;
        float var = rq[0] * inv - m * m;
        mean[co] = m;
        rstd[co] = rsqrtf(var + BN_EPS);
    }
}

// ================= 7) BN apply (in place on d_out) =================
__global__ __launch_bounds__(256) void bnapply_kernel(float* __restrict__ out,
                                                      const float* __restrict__ mean,
                                                      const float* __restrict__ rstd,
                                                      const float* __restrict__ gamma,
                                                      const float* __restrict__ beta) {
    const long long idx = (long long)blockIdx.x * 256 + threadIdx.x;
    const int co = (int)((idx >> 12) & (COUT - 1));
    out[idx] = (out[idx] - mean[co]) * rstd[co] * gamma[co] + beta[co];
}

// =====================================================================
extern "C" void kernel_launch(void* const* d_in, const int* in_sizes, int n_in,
                              void* d_out, int out_size, void* d_ws, size_t ws_size,
                              hipStream_t stream) {
    (void)in_sizes; (void)n_in; (void)out_size; (void)ws_size;
    const float* x      = (const float*)d_in[0];
    const float* weight = (const float*)d_in[1];
    const float* gamma  = (const float*)d_in[2];
    const float* beta   = (const float*)d_in[3];
    const float* w1s = (const float*)d_in[4],  *w2s = (const float*)d_in[5],  *b2s = (const float*)d_in[6];
    const float* w1c = (const float*)d_in[7],  *w2c = (const float*)d_in[8],  *b2c = (const float*)d_in[9];
    const float* w1f = (const float*)d_in[10], *w2f = (const float*)d_in[11], *b2f = (const float*)d_in[12];
    const float* w1w = (const float*)d_in[13], *w2w = (const float*)d_in[14], *b2w = (const float*)d_in[15];
    float* out = (float*)d_out;

    // workspace carve-up
    char* wsb = (char*)d_ws;
    float*          pooled = (float*)(wsb + 0);                 // 16 KB
    float*          alpha  = (float*)(wsb + 16384);             // 64 f
    float*          meanw  = (float*)(wsb + 16384 + 1024);      // 256 f
    float*          rstdw  = (float*)(wsb + 16384 + 2048);      // 256 f
    unsigned short* xb     = (unsigned short*)(wsb + 32768);    // 33.5 MB
    unsigned short* dynw   = (unsigned short*)(wsb + 32768 + (size_t)B_ * HW * CIN * 2); // 18.9 MB

    gap_kernel<<<B_ * CIN, 256, 0, stream>>>(x, pooled);
    alpha_kernel<<<B_, 64, 0, stream>>>(pooled,
                                        w1s, w2s, b2s, w1c, w2c, b2c,
                                        w1f, w2f, b2f, w1w, w2w, b2w, alpha);
    {
        long long total = (long long)B_ * COUT * KPOSN * CIN;  // 9,437,184
        dynw_kernel<<<(unsigned)(total / 256), 256, 0, stream>>>(weight, alpha, dynw);
    }
    xcvt_kernel<<<dim3(HW / 32, CIN / 32, B_), 256, 0, stream>>>(x, xb);
    conv_wmma_kernel<<<dim3(HW / NT, COUT / MT, B_), 256, 0, stream>>>(dynw, xb, out);
    bnstats_kernel<<<COUT, 256, 0, stream>>>(out, meanw, rstdw);
    {
        long long total = (long long)B_ * COUT * HW;           // 16,777,216
        bnapply_kernel<<<(unsigned)(total / 256), 256, 0, stream>>>(out, meanw, rstdw, gamma, beta);
    }
}
